// GlobalModel_79319456022825
// MI455X (gfx1250) — compile-verified, matrix operand order checked by reading, not run on hardware
//
#include <hip/hip_runtime.h>
#include <hip/hip_bf16.h>
#include <stdint.h>

typedef __bf16 v16bf __attribute__((ext_vector_type(16)));
typedef short  v16s  __attribute__((ext_vector_type(16)));
typedef int    v8i   __attribute__((ext_vector_type(8)));
typedef float  v8f   __attribute__((ext_vector_type(8)));

#define NN   1000000
#define BB   16384
#define FN   128
#define DIN  256
#define HA   64
#define HG   256
#define FO   128
#define MAXSEG 1024

union V16U { v16s s; v16bf b; v8i i; };

__device__ __forceinline__ unsigned short f2bf(float f) {
  union { float f; unsigned u; } v; v.f = f;
  unsigned u = v.u;
  unsigned r = u + 0x7FFFu + ((u >> 16) & 1u);
  return (unsigned short)(r >> 16);
}
// pack two floats -> two bf16 in one dword: 2 adds + 1 v_perm_b32
__device__ __forceinline__ unsigned pk2bf(float a, float b) {
  union { float f; unsigned u; } ua, ub; ua.f = a; ub.f = b;
  return __builtin_amdgcn_perm(ub.u + 0x8000u, ua.u + 0x8000u, 0x07060302u);
}
// fill a 16-element bf16 A/B fragment from two 8-float runs at src, src+16
__device__ __forceinline__ void fill_frag(V16U& fr, const float* src) {
  float4 a0 = *(const float4*)(src);
  float4 a1 = *(const float4*)(src + 4);
  float4 a2 = *(const float4*)(src + 16);
  float4 a3 = *(const float4*)(src + 20);
  fr.i[0] = pk2bf(a0.x, a0.y); fr.i[1] = pk2bf(a0.z, a0.w);
  fr.i[2] = pk2bf(a1.x, a1.y); fr.i[3] = pk2bf(a1.z, a1.w);
  fr.i[4] = pk2bf(a2.x, a2.y); fr.i[5] = pk2bf(a2.z, a2.w);
  fr.i[6] = pk2bf(a3.x, a3.y); fr.i[7] = pk2bf(a3.z, a3.w);
}

// ---- pack weight matrix W[K x Nf] (f32 row-major) into bf16 WMMA B-fragment order
__global__ void pack_kernel(const float* __restrict__ W, unsigned short* __restrict__ out,
                            int K, int Nf) {
  int idx = blockIdx.x * blockDim.x + threadIdx.x;
  if (idx >= K * Nf) return;
  int tile = idx >> 9;
  int rem  = idx & 511;
  int lane = rem >> 4;
  int j    = rem & 15;
  int ntn  = Nf >> 4;
  int kc = tile / ntn, nt = tile % ntn;
  int n  = nt * 16 + (lane & 15);
  int k0 = kc * 32 + ((lane >> 4) ? 8 : 0);
  int k  = (j < 8) ? (k0 + j) : (k0 + 16 + (j - 8));
  out[idx] = f2bf(W[(long)k * Nf + n]);
}

// ---- fused per-segment kernel: one wave per graph.
// scores (WMMA MLP) -> local softmax (shuffle/LDS, no atomics) -> attn + normalized pooling.
__global__ __launch_bounds__(128) void segment_kernel(
    const float* __restrict__ x, const float* __restrict__ u, const int* __restrict__ batch,
    const unsigned short* __restrict__ w1p, const float* __restrict__ b1,
    const float* __restrict__ w2, const float* __restrict__ b2,
    float* __restrict__ pooled, float* __restrict__ attn)
{
  __shared__ float w2s[HA];
  __shared__ float b1s[HA];
  __shared__ float hbuf[4][16][68];       // +4 pad kills bank conflicts
  __shared__ float sbuf[4][MAXSEG];
  int tid = threadIdx.x;
  if (tid < HA) { w2s[tid] = w2[tid]; b1s[tid] = b1[tid]; }
  __syncthreads();

  int wave = tid >> 5, lane = tid & 31;
  int b = blockIdx.x * 4 + wave;          // one graph per wave
  int m = lane & 15, half = lane >> 4;

  // segment bounds in sorted batch[] (uniform across wave)
  int lo = 0, hi = NN;
  while (lo < hi) { int mid = (lo + hi) >> 1; if (batch[mid] < b) lo = mid + 1; else hi = mid; }
  int start = lo;
  int lo2 = start, hi2 = NN;
  while (lo2 < hi2) { int mid = (lo2 + hi2) >> 1; if (batch[mid] < b + 1) lo2 = mid + 1; else hi2 = mid; }
  int cnt = lo2 - start;
  if (cnt > MAXSEG) cnt = MAXSEG;         // ~impossible for Poisson(61)

  long prow = (long)b * FN + lane * 4;
  if (cnt == 0) {                          // empty graph: pooled row = 0
    float4 z = {0.f, 0.f, 0.f, 0.f};
    *(float4*)(pooled + prow) = z;
    return;                                // wave-uniform exit
  }

  // u-half of A fragments (K = 128..255) is constant for the whole segment
  const float* ur = u + (long)b * 128;
  V16U au[4];
  #pragma unroll
  for (int kc = 0; kc < 4; ++kc) fill_frag(au[kc], ur + kc * 32 + half * 8);

  int ntile = (cnt + 15) >> 4;
  float lmax = -3.0e38f;
  for (int t = 0; t < ntile; ++t) {
    int ridx = t * 16 + m;
    long node = start + ((ridx < cnt) ? ridx : 0);   // clamp: extra rows discarded
    const float* xr = x + node * FN;
    V16U ax[4];
    #pragma unroll
    for (int kc = 0; kc < 4; ++kc) fill_frag(ax[kc], xr + kc * 32 + half * 8);

    for (int nt = 0; nt < 4; ++nt) {
      v8f c = {0.f,0.f,0.f,0.f,0.f,0.f,0.f,0.f};
      #pragma unroll
      for (int kc = 0; kc < 4; ++kc) {
        V16U bf; bf.s = *(const v16s*)(w1p + (((kc * 4 + nt) * 32 + lane) << 4));
        c = __builtin_amdgcn_wmma_f32_16x16x32_bf16(false, ax[kc].b, false, bf.b,
                                                    (short)0, c, false, false);
      }
      #pragma unroll
      for (int kc = 0; kc < 4; ++kc) {
        V16U bf; bf.s = *(const v16s*)(w1p + ((((kc + 4) * 4 + nt) * 32 + lane) << 4));
        c = __builtin_amdgcn_wmma_f32_16x16x32_bf16(false, au[kc].b, false, bf.b,
                                                    (short)0, c, false, false);
      }
      int col = nt * 16 + m;
      #pragma unroll
      for (int r = 0; r < 8; ++r) {
        float v = c[r] + b1s[col];
        hbuf[wave][half * 8 + r][col] = v > 0.f ? v : 0.f;
      }
    }
    if (lane < 16) {
      float s = b2[0];
      #pragma unroll 8
      for (int cc = 0; cc < HA; ++cc) s += hbuf[wave][lane][cc] * w2s[cc];
      int idx = t * 16 + lane;
      if (idx < cnt) { sbuf[wave][idx] = s; lmax = fmaxf(lmax, s); }
    }
  }

  // wave-wide max
  #pragma unroll
  for (int o = 16; o >= 1; o >>= 1) lmax = fmaxf(lmax, __shfl_xor(lmax, o, 32));
  // exp + wave-wide sum
  float lsum = 0.f;
  for (int i = lane; i < cnt; i += 32) {
    float e = __expf(sbuf[wave][i] - lmax);
    sbuf[wave][i] = e;
    lsum += e;
  }
  #pragma unroll
  for (int o = 16; o >= 1; o >>= 1) lsum += __shfl_xor(lsum, o, 32);
  float invd = 1.f / lsum;

  // attn output (coalesced)
  for (int i = lane; i < cnt; i += 32) attn[start + i] = sbuf[wave][i] * invd;

  // normalized weighted pooling; x rows are L2-resident from the score pass
  float4 acc = {0.f, 0.f, 0.f, 0.f};
  for (int i = 0; i < cnt; ++i) {
    float e = sbuf[wave][i];
    float4 xv = *(const float4*)(x + (long)(start + i) * FN + lane * 4);
    acc.x += xv.x * e; acc.y += xv.y * e; acc.z += xv.z * e; acc.w += xv.w * e;
  }
  acc.x *= invd; acc.y *= invd; acc.z *= invd; acc.w *= invd;
  *(float4*)(pooled + prow) = acc;
}

// ---- global MLP: out = relu([u | pooled] @ Wg1 + bg1) @ Wg2 + bg2  (pooled pre-normalized)
__global__ __launch_bounds__(128) void gmlp_kernel(
    const float* __restrict__ u, const float* __restrict__ pooled,
    const unsigned short* __restrict__ wg1p, const float* __restrict__ bg1,
    const unsigned short* __restrict__ wg2p, const float* __restrict__ bg2,
    float* __restrict__ out)
{
  __shared__ float bg1s[HG];
  __shared__ float bg2s[FO];
  __shared__ float hst[4][16][HG + 4];
  int tid = threadIdx.x;
  for (int i = tid; i < HG; i += 128) bg1s[i] = bg1[i];
  if (tid < FO) bg2s[tid] = bg2[tid];
  __syncthreads();

  int wave = tid >> 5, lane = tid & 31;
  int wtile = blockIdx.x * 4 + wave;
  int m = lane & 15, half = lane >> 4;
  long row = (long)wtile * 16 + m;
  const float* ur = u + row * 128;
  const float* pr = pooled + row * FN;

  V16U afr[8];
  #pragma unroll
  for (int kc = 0; kc < 8; ++kc) {
    int k0 = kc * 32 + half * 8;
    const float* src = (kc < 4) ? (ur + k0) : (pr + (k0 - 128));
    fill_frag(afr[kc], src);
  }

  for (int nt = 0; nt < 16; ++nt) {
    v8f c = {0.f,0.f,0.f,0.f,0.f,0.f,0.f,0.f};
    #pragma unroll
    for (int kc = 0; kc < 8; ++kc) {
      V16U bf; bf.s = *(const v16s*)(wg1p + (((kc * 16 + nt) * 32 + lane) << 4));
      c = __builtin_amdgcn_wmma_f32_16x16x32_bf16(false, afr[kc].b, false, bf.b,
                                                  (short)0, c, false, false);
    }
    int col = nt * 16 + m;
    #pragma unroll
    for (int r = 0; r < 8; ++r) {
      float v = c[r] + bg1s[col];
      hst[wave][half * 8 + r][col] = v > 0.f ? v : 0.f;
    }
  }

  V16U af2[8];
  #pragma unroll
  for (int kc = 0; kc < 8; ++kc) fill_frag(af2[kc], &hst[wave][m][kc * 32 + half * 8]);

  for (int nt = 0; nt < 8; ++nt) {
    v8f c = {0.f,0.f,0.f,0.f,0.f,0.f,0.f,0.f};
    #pragma unroll
    for (int kc = 0; kc < 8; ++kc) {
      V16U bf; bf.s = *(const v16s*)(wg2p + (((kc * 8 + nt) * 32 + lane) << 4));
      c = __builtin_amdgcn_wmma_f32_16x16x32_bf16(false, af2[kc].b, false, bf.b,
                                                  (short)0, c, false, false);
    }
    int col = nt * 16 + m;
    #pragma unroll
    for (int r = 0; r < 8; ++r) {
      long orow = (long)wtile * 16 + half * 8 + r;
      out[orow * FO + col] = c[r] + bg2s[col];
    }
  }
}

extern "C" void kernel_launch(void* const* d_in, const int* in_sizes, int n_in,
                              void* d_out, int out_size, void* d_ws, size_t ws_size,
                              hipStream_t stream)
{
  (void)in_sizes; (void)n_in; (void)out_size; (void)ws_size;
  const float* x    = (const float*)d_in[0];
  const float* u    = (const float*)d_in[3];
  const int*   batch= (const int*)  d_in[4];
  const float* W1   = (const float*)d_in[5];
  const float* b1   = (const float*)d_in[6];
  const float* W2   = (const float*)d_in[7];
  const float* b2   = (const float*)d_in[8];
  const float* Wg1  = (const float*)d_in[9];
  const float* bg1  = (const float*)d_in[10];
  const float* Wg2  = (const float*)d_in[11];
  const float* bg2  = (const float*)d_in[12];

  float* out  = (float*)d_out;                 // [B*FO]
  float* attn = out + (size_t)BB * FO;         // [N]

  char* ws = (char*)d_ws;
  size_t off = 0;
  auto take = [&](size_t bytes) -> char* {
    char* p = ws + off;
    off = (off + bytes + 255) & ~(size_t)255;
    return p;
  };
  unsigned short* w1p  = (unsigned short*)take((size_t)DIN * HA * 2);
  unsigned short* wg1p = (unsigned short*)take((size_t)DIN * HG * 2);
  unsigned short* wg2p = (unsigned short*)take((size_t)HG * FO * 2);
  float*          pooled = (float*)take((size_t)BB * FN * 4);

  // 1) pack weights into bf16 fragment order (once per call; tiny)
  pack_kernel<<<(DIN * HA + 255) / 256, 256, 0, stream>>>(W1,  w1p,  DIN, HA);
  pack_kernel<<<(DIN * HG + 255) / 256, 256, 0, stream>>>(Wg1, wg1p, DIN, HG);
  pack_kernel<<<(HG  * FO + 255) / 256, 256, 0, stream>>>(Wg2, wg2p, HG,  FO);

  // 2) fused per-graph: score GEMM + softmax + attn + normalized pooling (single x pass to HBM)
  segment_kernel<<<BB / 4, 128, 0, stream>>>(x, u, batch, w1p, b1, W2, b2, pooled, attn);

  // 3) global MLP on WMMA
  gmlp_kernel<<<(BB / 16) / 4, 128, 0, stream>>>(u, pooled, wg1p, bg1, wg2p, bg2, out);
}
